// PPFEmbeddingSin_25615184953560
// MI455X (gfx1250) — matrix operand order repaired and checked by reading, not compiled, not run on hardware
//
#include <hip/hip_runtime.h>
#include <hip/hip_bf16.h>

// ---- problem constants (from reference setup_inputs) ----
#define BB   2
#define NN   16384
#define MM_  384
#define KK   128
#define DD   256
#define DD3  768
#define MM2  (MM_ * MM_)      // 147456 pairs per batch
#define TPB  256
#define FREQ_C  (-0.07195578429166302f)   // -ln(10000)/128
#define TWO_PI  6.283185307179586f

typedef __attribute__((ext_vector_type(16))) _Float16 v16h;
typedef __attribute__((ext_vector_type(8)))  float    v8f;

__device__ __forceinline__ float angle3(float ax, float ay, float az,
                                        float bx, float by, float bz) {
    float cx = ay * bz - az * by;
    float cy = az * bx - ax * bz;
    float cz = ax * by - ay * bx;
    float s = sqrtf(cx * cx + cy * cy + cz * cz);
    float c = ax * bx + ay * by + az * bz;
    return atan2f(s, c);
}

// ---------------------------------------------------------------------------
// Kernel 0: convert W_d (256x256) and W_patch (768x256) to f16, TRANSPOSED:
//   WdhT[n*256 + k] = W_d[k*256 + n]     (n = output col, k = reduction dim)
//   WphT[n*768 + k] = W_patch[k*256 + n]
// This makes each WMMA B-fragment a single contiguous 32B load.
// ---------------------------------------------------------------------------
__global__ void k_cvt_weights(const float* __restrict__ Wd,
                              const float* __restrict__ Wp,
                              _Float16* __restrict__ WdhT,
                              _Float16* __restrict__ WphT) {
    int i = blockIdx.x * TPB + threadIdx.x;
    if (i < DD * DD) {
        int k = i >> 8, n = i & 255;
        WdhT[(size_t)n * DD + k] = (_Float16)Wd[i];
    }
    if (i < DD3 * DD) {
        int k = i >> 8, n = i & 255;
        WphT[(size_t)n * DD3 + k] = (_Float16)Wp[i];
    }
}

// ---------------------------------------------------------------------------
// Kernel 1: assign each point to its nearest node (argmin over M=384)
// ---------------------------------------------------------------------------
__global__ void k_assign(const float* __restrict__ points,
                         const float* __restrict__ nodes,
                         int* __restrict__ p2n) {
    __shared__ float sn[MM_ * 3];
    const int bpb = NN / TPB;
    int b = blockIdx.x / bpb;
    int n = (blockIdx.x % bpb) * TPB + threadIdx.x;
    for (int i = threadIdx.x; i < MM_ * 3; i += TPB)
        sn[i] = nodes[(size_t)b * MM_ * 3 + i];
    __syncthreads();
    const float* p = points + ((size_t)b * NN + n) * 3;
    float px = p[0], py = p[1], pz = p[2];
    float best = 3.4e38f;
    int   bi = 0;
    for (int m = 0; m < MM_; ++m) {
        float dx = sn[m * 3 + 0] - px;
        float dy = sn[m * 3 + 1] - py;
        float dz = sn[m * 3 + 2] - pz;
        float d = dx * dx + dy * dy + dz * dz;
        if (d < best) { best = d; bi = m; }
    }
    p2n[(size_t)b * NN + n] = bi;
}

// ---------------------------------------------------------------------------
// Kernel 2: per-node K=128 nearest of its assigned points.
// One block per (b, m). Distances as f16 in LDS (32KB); 128 extraction
// passes, each a wave32 shuffle argmin + 8-wave LDS combine.
// ---------------------------------------------------------------------------
__global__ void k_knn(const float* __restrict__ points,
                      const float* __restrict__ nodes,
                      const int* __restrict__ p2n,
                      int* __restrict__ knn_idx) {
    __shared__ _Float16 dist[NN];   // 32 KB
    __shared__ float rv[8];
    __shared__ int   ri[8];
    int b = blockIdx.x / MM_;
    int m = blockIdx.x % MM_;
    const float* nd = nodes + ((size_t)b * MM_ + m) * 3;
    float nx = nd[0], ny = nd[1], nz = nd[2];
    int t = threadIdx.x;
    int lane = t & 31, w = t >> 5;

    for (int n = t; n < NN; n += TPB) {
        const float* p = points + ((size_t)b * NN + n) * 3;
        float dx = p[0] - nx, dy = p[1] - ny, dz = p[2] - nz;
        float d = dx * dx + dy * dy + dz * dz;
        bool mine = (p2n[(size_t)b * NN + n] == m);
        dist[n] = mine ? (_Float16)d : (_Float16)60000.0f;   // masked = 1e12 analog
    }
    __syncthreads();

    for (int k = 0; k < KK; ++k) {
        float bv = 1e30f;
        int   bi = 0;
        for (int n = t; n < NN; n += TPB) {
            float d = (float)dist[n];
            if (d < bv) { bv = d; bi = n; }
        }
        // wave32 argmin reduce
        for (int off = 16; off > 0; off >>= 1) {
            float ov = __shfl_xor(bv, off);
            int   oi = __shfl_xor(bi, off);
            if (ov < bv) { bv = ov; bi = oi; }
        }
        if (lane == 0) { rv[w] = bv; ri[w] = bi; }
        __syncthreads();
        if (t == 0) {
            float fb = rv[0]; int fi = ri[0];
            for (int i = 1; i < 8; ++i)
                if (rv[i] < fb) { fb = rv[i]; fi = ri[i]; }
            knn_idx[((size_t)b * MM_ + m) * KK + k] = fi;
            dist[fi] = (_Float16)65000.0f;   // remove from candidate set
        }
        __syncthreads();
    }
}

// ---------------------------------------------------------------------------
// Kernel 3: global embedding GEMM.
// out[b, p, :] = (sin_embed(d) + sin_embed(a_ij)) @ W_d + 2*b_d   (p = i*M+j)
// One block = 16 pair-rows x 256 cols. 8 waves x 2 col-tiles,
// K=256 -> 8 x v_wmma_f32_16x16x32_f16 per tile. B frags = contiguous v16h.
// ---------------------------------------------------------------------------
__global__ void k_global_emb(const float* __restrict__ nodes,
                             const float* __restrict__ nnorm,
                             const _Float16* __restrict__ WdhT,
                             const float* __restrict__ bd,
                             float* __restrict__ out) {
    __shared__ _Float16 Ah[16][264];   // 264 halves = 528B row stride (16B aligned, conflict-free)
    const int tilesPerBatch = MM2 / 16;
    int b    = blockIdx.x / tilesPerBatch;
    int tile = blockIdx.x % tilesPerBatch;
    int row0 = tile * 16;

    int t  = threadIdx.x;
    int r  = t >> 4;
    int c0 = (t & 15) * 16;
    int p  = row0 + r;
    int i  = p / MM_;
    int j  = p - i * MM_;
    const float* Ni  = nodes + ((size_t)b * MM_ + i) * 3;
    const float* Nj  = nodes + ((size_t)b * MM_ + j) * 3;
    const float* NNi = nnorm + ((size_t)b * MM_ + i) * 3;
    float dx = Ni[0] - Nj[0], dy = Ni[1] - Nj[1], dz = Ni[2] - Nj[2];
    float dval = sqrtf(dx * dx + dy * dy + dz * dz) * TWO_PI;
    float aij  = angle3(dx, dy, dz, NNi[0], NNi[1], NNi[2]);

    #pragma unroll
    for (int q = 0; q < 16; ++q) {
        int k = c0 + q;
        int f = k >> 1;
        float dt  = __expf((float)f * FREQ_C);
        float om1 = dval * dt, om2 = aij * dt;
        float v = (k & 1) ? (__cosf(om1) + __cosf(om2))
                          : (__sinf(om1) + __sinf(om2));
        Ah[r][k] = (_Float16)v;
    }
    __syncthreads();

    int lane = t & 31, w = t >> 5;
    int hi = lane >> 4, lo = lane & 15;
    int n0 = w * 32 + lo;
    int n1 = n0 + 16;
    const _Float16* B0 = WdhT + (size_t)n0 * DD;   // column n0's K-values, contiguous
    const _Float16* B1 = WdhT + (size_t)n1 * DD;
    v8f acc0 = {}, acc1 = {};

    #pragma unroll
    for (int kk = 0; kk < 8; ++kk) {
        int Kb = kk * 32 + 16 * hi;                // B frag: 16 contiguous K halves
        v16h b0 = *(const v16h*)(B0 + Kb);
        v16h b1 = *(const v16h*)(B1 + Kb);
        v16h a;
        #pragma unroll
        for (int v = 0; v < 8; ++v) {
            int Ka = kk * 32 + 16 * hi + 2 * v;   // A frag: 16 contiguous K halves (matches B)
            a[2 * v]     = Ah[lo][Ka];
            a[2 * v + 1] = Ah[lo][Ka + 1];
        }
        acc0 = __builtin_amdgcn_wmma_f32_16x16x32_f16(false, a, false, b0, (short)0, acc0, false, false);
        acc1 = __builtin_amdgcn_wmma_f32_16x16x32_f16(false, a, false, b1, (short)0, acc1, false, false);
    }

    float bias0 = 2.0f * bd[n0];
    float bias1 = 2.0f * bd[n1];
    float* obase = out + ((size_t)b * MM2 + row0) * DD;
    #pragma unroll
    for (int v = 0; v < 8; ++v) {
        int mrow = v + 8 * hi;
        obase[(size_t)mrow * DD + n0] = acc0[v] + bias0;
        obase[(size_t)mrow * DD + n1] = acc1[v] + bias1;
    }
}

// ---------------------------------------------------------------------------
// Kernel 4: patch GEMM + fused max-over-K.
// One block per (b, m): 8 row-tiles of 16 knn rows, A = 16x768 f16 in LDS,
// 24 WMMAs per output tile, running f32 max in registers, half-wave
// __shfl_xor(16) combine at the end (wave32).
// ---------------------------------------------------------------------------
__global__ void k_patch(const float* __restrict__ points,
                        const float* __restrict__ pnorm,
                        const float* __restrict__ nodes,
                        const float* __restrict__ nnorm,
                        const int* __restrict__ knn_idx,
                        const _Float16* __restrict__ WphT,
                        const float* __restrict__ bp,
                        float* __restrict__ local_out) {
    __shared__ _Float16 Ah[16][776];   // 768 + 8 pad -> 1552B row stride (16B aligned)
    int b = blockIdx.x / MM_;
    int m = blockIdx.x % MM_;
    const float* nd = nodes + ((size_t)b * MM_ + m) * 3;
    const float* nv = nnorm + ((size_t)b * MM_ + m) * 3;
    float nmx = nd[0], nmy = nd[1], nmz = nd[2];
    float nnx = nv[0], nny = nv[1], nnz = nv[2];

    int t = threadIdx.x;
    int lane = t & 31, w = t >> 5;
    int hi = lane >> 4, lo = lane & 15;
    int r  = t >> 4;
    int c0 = (t & 15) * 48;
    int n0 = w * 32 + lo;
    int n1 = n0 + 16;
    const _Float16* B0 = WphT + (size_t)n0 * DD3;
    const _Float16* B1 = WphT + (size_t)n1 * DD3;

    v8f mx0, mx1;
    #pragma unroll
    for (int v = 0; v < 8; ++v) { mx0[v] = -3.4e38f; mx1[v] = -3.4e38f; }

    for (int rt = 0; rt < 8; ++rt) {
        int krow = rt * 16 + r;
        int pidx = knn_idx[((size_t)b * MM_ + m) * KK + krow];
        const float* P  = points + ((size_t)b * NN + pidx) * 3;
        const float* PN = pnorm  + ((size_t)b * NN + pidx) * 3;
        float vx = P[0] - nmx, vy = P[1] - nmy, vz = P[2] - nmz;
        float a1 = angle3(vx, vy, vz, nnx, nny, nnz);
        float a2 = angle3(-vx, -vy, -vz, PN[0], PN[1], PN[2]);
        float a3 = angle3(nnx, nny, nnz, PN[0], PN[1], PN[2]);

        for (int q = 0; q < 48; ++q) {
            int c = c0 + q;
            int sect = c >> 8;
            int k = c & 255;
            float x = (sect == 0) ? a1 : ((sect == 1) ? a2 : a3);
            int f = k >> 1;
            float om = x * __expf((float)f * FREQ_C);
            Ah[r][c] = (_Float16)((k & 1) ? __cosf(om) : __sinf(om));
        }
        __syncthreads();

        v8f acc0 = {}, acc1 = {};
        for (int kk = 0; kk < 24; ++kk) {
            int Kb = kk * 32 + 16 * hi;
            v16h b0 = *(const v16h*)(B0 + Kb);
            v16h b1 = *(const v16h*)(B1 + Kb);
            v16h a;
            #pragma unroll
            for (int v = 0; v < 8; ++v) {
                int Ka = kk * 32 + 16 * hi + 2 * v;
                a[2 * v]     = Ah[lo][Ka];
                a[2 * v + 1] = Ah[lo][Ka + 1];
            }
            acc0 = __builtin_amdgcn_wmma_f32_16x16x32_f16(false, a, false, b0, (short)0, acc0, false, false);
            acc1 = __builtin_amdgcn_wmma_f32_16x16x32_f16(false, a, false, b1, (short)0, acc1, false, false);
        }
        #pragma unroll
        for (int v = 0; v < 8; ++v) {
            mx0[v] = fmaxf(mx0[v], acc0[v]);
            mx1[v] = fmaxf(mx1[v], acc1[v]);
        }
        __syncthreads();
    }

    float m0 = -3.4e38f, m1 = -3.4e38f;
    #pragma unroll
    for (int v = 0; v < 8; ++v) { m0 = fmaxf(m0, mx0[v]); m1 = fmaxf(m1, mx1[v]); }
    m0 = fmaxf(m0, __shfl_xor(m0, 16));   // combine row halves (wave32)
    m1 = fmaxf(m1, __shfl_xor(m1, 16));
    if (hi == 0) {
        float* lo_out = local_out + ((size_t)b * MM_ + m) * DD;
        lo_out[n0] = m0 + bp[n0];
        lo_out[n1] = m1 + bp[n1];
    }
}

// ---------------------------------------------------------------------------
extern "C" void kernel_launch(void* const* d_in, const int* in_sizes, int n_in,
                              void* d_out, int out_size, void* d_ws, size_t ws_size,
                              hipStream_t stream) {
    const float* points = (const float*)d_in[0];
    const float* nodes  = (const float*)d_in[1];
    const float* pnorm  = (const float*)d_in[2];
    const float* nnorm  = (const float*)d_in[3];
    const float* Wd     = (const float*)d_in[4];
    const float* bd     = (const float*)d_in[5];
    const float* Wp     = (const float*)d_in[6];
    const float* bp     = (const float*)d_in[7];
    float* out = (float*)d_out;

    char* ws = (char*)d_ws;
    int*      p2n  = (int*)ws;                                   // B*N*4        = 128 KB
    int*      knn  = (int*)(ws + (size_t)BB * NN * 4);           // B*M*K*4      = 384 KB
    _Float16* WdhT = (_Float16*)(ws + (size_t)BB * NN * 4
                                    + (size_t)BB * MM_ * KK * 4);             // 128 KB
    _Float16* WphT = WdhT + DD * DD;                                          // 384 KB

    k_cvt_weights<<<(DD3 * DD + TPB - 1) / TPB, TPB, 0, stream>>>(Wd, Wp, WdhT, WphT);
    k_assign<<<(BB * NN) / TPB, TPB, 0, stream>>>(points, nodes, p2n);
    k_knn<<<BB * MM_, TPB, 0, stream>>>(points, nodes, p2n, knn);
    k_global_emb<<<BB * (MM2 / 16), TPB, 0, stream>>>(nodes, nnorm, WdhT, bd, out);
    k_patch<<<BB * MM_, TPB, 0, stream>>>(points, pnorm, nodes, nnorm, knn, WphT, bp,
                                          out + (size_t)BB * MM2 * DD);
}
